// Attention_1073741824296
// MI455X (gfx1250) — matrix operand order repaired and checked by reading, not compiled, not run on hardware
//
#include <hip/hip_runtime.h>
#include <math.h>

static constexpr int Bsz = 4;
static constexpr int T   = 4096;
static constexpr int C   = 1024;
static constexpr int H   = 64;

typedef _Float16     v16h __attribute__((ext_vector_type(16)));
typedef float        v8f  __attribute__((ext_vector_type(8)));
typedef unsigned int v4u  __attribute__((ext_vector_type(4)));

union HFrag { v16h h; v4u q[2]; unsigned int u[8]; _Float16 e[16]; };

// LDS byte offset of a shared-memory pointer (flat LDS aperture: low 32 bits
// are the workgroup-relative LDS address).
__device__ inline unsigned lds_off(const void* p) {
  return (unsigned)(unsigned long long)p;
}

// A-matrix fragment (16x32 f16). Per ISA layout, lane m needs K = 8*hi+[0..7]
// and K = 16+8*hi+[0..7] -> two 16B loads from a K-contiguous row.
__device__ inline v16h load_afrag(const _Float16* rowbase, int hi) {
  HFrag f;
  f.q[0] = *reinterpret_cast<const v4u*>(rowbase + (hi << 3));
  f.q[1] = *reinterpret_cast<const v4u*>(rowbase + 16 + (hi << 3));
  return f.h;
}

// B-matrix fragment (32x16 f16) from a source stored transposed
// (row n holds K contiguously): lane n needs K = 8*hi+[0..7] and
// K = 16+8*hi+[0..7] (+koff) -> two 16B loads.
__device__ inline v16h load_bfragT(const _Float16* rowbase, int hi, int koff) {
  HFrag f;
  f.q[0] = *reinterpret_cast<const v4u*>(rowbase + koff + (hi << 3));
  f.q[1] = *reinterpret_cast<const v4u*>(rowbase + koff + 16 + (hi << 3));
  return f.h;
}

// Async 16B global->LDS copy (CDNA5), GVS mode: lds byte addr in VGPR,
// 32-bit byte offset in VGPR, 64-bit base in SGPRs. Tracked by ASYNCcnt.
__device__ inline void async_g2l_b128(unsigned lds_addr, unsigned goff, const void* base) {
  asm volatile("global_load_async_to_lds_b128 %0, %1, %2 offset:0"
               :: "v"(lds_addr), "v"(goff), "s"(base) : "memory");
}

// Transpose-convert weights: dst[h*C + c] = (f16)src[c*H + h]
__global__ void wconvT_kernel(const float* __restrict__ src, _Float16* __restrict__ dst) {
  int i = blockIdx.x * blockDim.x + threadIdx.x;   // over H*C
  int h = i >> 10;          // /C
  int c = i & (C - 1);
  dst[i] = (_Float16)src[c * H + h];
}

// q/k/v projection: block = 4 waves = 64 rows of x; each wave a 16-row tile.
// Weights pre-transposed [mtx][H][C]; chunk staged to LDS via async copy.
// v written transposed [B][H][T] for the attention kernel's P*V B-frags.
__global__ void __launch_bounds__(128)
proj_kernel(const float* __restrict__ x, const _Float16* __restrict__ wT16,
            _Float16* __restrict__ q16, _Float16* __restrict__ k16,
            _Float16* __restrict__ vT) {
  __shared__ __align__(16) _Float16 xs[64 * 32];        // x chunk [row][k]
  __shared__ __align__(16) _Float16 wshT[3 * 64 * 32];  // W chunk [mtx][h][k]

  const int tid  = threadIdx.x;
  const int wave = tid >> 5;
  const int lane = tid & 31;
  const int n    = lane & 15;
  const int hi   = (lane >> 4) & 1;
  const int rowbase = blockIdx.x * 64;

  const unsigned wsh_lds = lds_off(&wshT[0]);

  v8f acc[12] = {};

  for (int kc = 0; kc < C; kc += 32) {
    // Async stage of transposed-weight chunk: 3 mtx x 64 h-rows x 16 K halfs
    // = 384 x 16B segments, 3 per thread.
#pragma unroll
    for (int j = 0; j < 3; ++j) {
      int s   = tid + j * 128;           // 0..383
      int mtx = s >> 7;
      int rem = s & 127;
      int h = rem >> 1, seg = rem & 1;
      unsigned go = (unsigned)((((mtx * H + h) * C) + kc + seg * 8) * 2);
      unsigned la = wsh_lds + (unsigned)((mtx * 2048 + h * 32 + seg * 8) * 2);
      async_g2l_b128(la, go, wT16);
    }
    // x chunk: convert f32->f16, packed pair stores (overlaps with async).
    for (int i = tid; i < 1024; i += 128) {
      int r = i >> 4, cp = (i & 15) * 2;
      float2 f = *reinterpret_cast<const float2*>(&x[(size_t)(rowbase + r) * C + kc + cp]);
      union { _Float16 e[2]; unsigned int u; } pk;
      pk.e[0] = (_Float16)f.x; pk.e[1] = (_Float16)f.y;
      *reinterpret_cast<unsigned int*>(&xs[r * 32 + cp]) = pk.u;
    }
    asm volatile("s_wait_asynccnt 0x0" ::: "memory");
    __syncthreads();

    v16h a = load_afrag(&xs[(wave * 16 + n) * 32], hi);

#pragma unroll
    for (int mtx = 0; mtx < 3; ++mtx) {
#pragma unroll
      for (int nt = 0; nt < 4; ++nt) {
        v16h bf = load_bfragT(&wshT[(mtx * 64 + nt * 16 + n) * 32], hi, 0);
        acc[mtx * 4 + nt] = __builtin_amdgcn_wmma_f32_16x16x32_f16(
            false, a, false, bf, (short)0, acc[mtx * 4 + nt], false, false);
      }
    }
    __syncthreads();
  }

#pragma unroll
  for (int nt = 0; nt < 4; ++nt) {
#pragma unroll
    for (int r = 0; r < 8; ++r) {
      int row = rowbase + wave * 16 + r + 8 * hi;
      int col = nt * 16 + n;
      q16[(size_t)row * H + col] = (_Float16)acc[0 * 4 + nt][r];
      k16[(size_t)row * H + col] = (_Float16)acc[1 * 4 + nt][r];
      int bb = row >> 12;                 // /T
      int t  = row & (T - 1);
      vT[((size_t)bb * H + col) * T + t] = (_Float16)acc[2 * 4 + nt][r];
    }
  }
}

// Fused causal flash attention. Block = 4 waves covering 64 query rows.
// K/V tiles (32 keys) double-buffered in LDS via async copies, pipelined
// one block ahead of the WMMA/softmax compute. Softmax row-sums computed
// with a P*ones WMMA (lands replicated in exactly the per-lane stat slots).
__global__ void __launch_bounds__(128)
attn_kernel(const _Float16* __restrict__ q16, const _Float16* __restrict__ k16,
            const _Float16* __restrict__ vT, float* __restrict__ out) {
  __shared__ __align__(16) _Float16 kt[2][32 * 64];   // [buf][key][h]
  __shared__ __align__(16) _Float16 vtt[2][64 * 32];  // [buf][h][key]
  __shared__ __align__(16) _Float16 pl[4 * 16 * 32];  // per-wave P scratch

  const int tid  = threadIdx.x;
  const int wave = tid >> 5;
  const int lane = tid & 31;
  const int n    = lane & 15;
  const int hi   = (lane >> 4) & 1;

  // Longest-first scheduling: high query tiles (most causal key blocks)
  // launch first, interleaved across batches.
  const int tilesPerBatch = T / 64;
  const int b     = blockIdx.x % Bsz;
  const int qt    = tilesPerBatch - 1 - (blockIdx.x / Bsz);
  const int qwg   = qt * 64;
  const int qbase = qwg + wave * 16;

  // Q A-frags (16x64 -> two 16x32 chunks), loaded once (2x b128 each).
  const _Float16* qrow = q16 + (size_t)(b * T + qbase + n) * H;
  v16h aq0 = load_afrag(qrow, hi);
  v16h aq1 = load_afrag(qrow + 32, hi);

  // Constant all-ones B fragment for row-sum-by-WMMA.
  v16h bone;
  {
    HFrag f;
#pragma unroll
    for (int i = 0; i < 8; ++i) f.u[i] = 0x3C003C00u;  // {1.0h, 1.0h}
    bone = f.h;
  }

  v8f o[4] = {};
  float mrun[8], lrun[8];
#pragma unroll
  for (int r = 0; r < 8; ++r) { mrun[r] = -INFINITY; lrun[r] = 0.0f; }

  // Stage key-block kb into LDS buffer bufi: 4 async b128 ops per thread
  // (2 for K tile, 2 for transposed-V tile) = 4 per wave -> ASYNCcnt += 4.
  auto stage = [&](int kb, int bufi) {
    const int k0 = kb * 32;
    const _Float16* kbase = k16 + (size_t)(b * T + k0) * H;   // 4KB contiguous
    const _Float16* vbase = vT + (size_t)b * H * T + k0;      // 64 rows x 64B
    unsigned kt_lds = lds_off(&kt[bufi][0]);
    unsigned vt_lds = lds_off(&vtt[bufi][0]);
#pragma unroll
    for (int j = 0; j < 2; ++j) {
      int s = tid * 2 + j;                       // 0..255
      async_g2l_b128(kt_lds + (unsigned)(s * 16), (unsigned)(s * 16), kbase);
    }
#pragma unroll
    for (int j = 0; j < 2; ++j) {
      int s = tid * 2 + j;                       // 0..255
      int h = s >> 2, seg = s & 3;
      unsigned go = (unsigned)((h * T + seg * 8) * 2);
      unsigned la = vt_lds + (unsigned)((h * 32 + seg * 8) * 2);
      async_g2l_b128(la, go, vbase);
    }
  };

  const int nkb = (qwg >> 5) + 2;   // causal: keys < qwg + 64
  stage(0, 0);

  for (int kb = 0; kb < nkb; ++kb) {
    const int k0  = kb * 32;
    const _Float16* ktb = &kt[kb & 1][0];
    const _Float16* vtb = &vtt[kb & 1][0];

    // Pipeline: issue next block's tiles, then wait for current block only.
    if (kb + 1 < nkb) {
      stage(kb + 1, (kb + 1) & 1);
      asm volatile("s_wait_asynccnt 0x4" ::: "memory");
    } else {
      asm volatile("s_wait_asynccnt 0x0" ::: "memory");
    }
    __syncthreads();

    // S = Q K^T (16x32) as two 16x16 f32 tiles, K=64 chained over two WMMAs.
    float sc[2][8];
#pragma unroll
    for (int nt = 0; nt < 2; ++nt) {
      const _Float16* krow = &ktb[(nt * 16 + n) * 64];
      v16h b0 = load_bfragT(krow, hi, 0);
      v16h b1 = load_bfragT(krow, hi, 32);
      v8f s = {};
      s = __builtin_amdgcn_wmma_f32_16x16x32_f16(false, aq0, false, b0, (short)0, s, false, false);
      s = __builtin_amdgcn_wmma_f32_16x16x32_f16(false, aq1, false, b1, (short)0, s, false, false);
#pragma unroll
      for (int r = 0; r < 8; ++r) {
        int row = qbase + r + 8 * hi;
        int col = k0 + nt * 16 + n;
        float sv = s[r] * 0.125f;                  // 1/sqrt(H=64)
        sc[nt][r] = (col <= row) ? sv : -INFINITY; // causal mask, no divergence
      }
    }

    // Online softmax: row max via 4-step half-wave shuffle; exps; rescale O.
    float ph[2][8];
    float alpha[8];
#pragma unroll
    for (int r = 0; r < 8; ++r) {
      float rmax = fmaxf(sc[0][r], sc[1][r]);
#pragma unroll
      for (int off = 1; off < 16; off <<= 1)
        rmax = fmaxf(rmax, __shfl_xor(rmax, off, 32));
      float newm = fmaxf(mrun[r], rmax);
      alpha[r]   = __expf(mrun[r] - newm);
      ph[0][r]   = __expf(sc[0][r] - newm);
      ph[1][r]   = __expf(sc[1][r] - newm);
      mrun[r]    = newm;
#pragma unroll
      for (int t = 0; t < 4; ++t) o[t][r] *= alpha[r];
    }

    // P: C-layout -> A-layout via per-wave LDS scratch (LDS in-order per wave).
    _Float16* pw = &pl[wave * 512];
#pragma unroll
    for (int nt = 0; nt < 2; ++nt)
#pragma unroll
      for (int r = 0; r < 8; ++r)
        pw[(r + 8 * hi) * 32 + nt * 16 + n] = (_Float16)ph[nt][r];
    __builtin_amdgcn_fence(__ATOMIC_ACQ_REL, "wavefront");
    v16h ap = load_afrag(&pw[n * 32], hi);

    // Row sums in one WMMA: P (16x32) * ones (32x16); every output column
    // equals the row sum, so each lane's C slot r holds rowsum(row r+8*hi).
    {
      v8f rs = {};
      rs = __builtin_amdgcn_wmma_f32_16x16x32_f16(false, ap, false, bone, (short)0, rs, false, false);
#pragma unroll
      for (int r = 0; r < 8; ++r)
        lrun[r] = lrun[r] * alpha[r] + rs[r];
    }

    // O += P V: B-frags straight out of the transposed V tile (2x b128 each).
#pragma unroll
    for (int nt = 0; nt < 4; ++nt) {
      v16h bv = load_bfragT(&vtb[(nt * 16 + n) * 32], hi, 0);
      o[nt] = __builtin_amdgcn_wmma_f32_16x16x32_f16(false, ap, false, bv, (short)0, o[nt], false, false);
    }
    __syncthreads();
  }

  // Normalize and store f32 output [B,T,H].
#pragma unroll
  for (int nt = 0; nt < 4; ++nt)
#pragma unroll
    for (int r = 0; r < 8; ++r) {
      int row = qbase + r + 8 * hi;
      out[(size_t)(b * T + row) * H + nt * 16 + n] = o[nt][r] / lrun[r];
    }
}

extern "C" void kernel_launch(void* const* d_in, const int* in_sizes, int n_in,
                              void* d_out, int out_size, void* d_ws, size_t ws_size,
                              hipStream_t stream) {
  (void)in_sizes; (void)n_in; (void)out_size; (void)ws_size;
  const float* x  = (const float*)d_in[0];
  const float* Wq = (const float*)d_in[1];
  const float* Wk = (const float*)d_in[2];
  const float* Wv = (const float*)d_in[3];
  float* out = (float*)d_out;

  // Workspace (halfs): [WqT|WkT|WvT (3*C*H)][q16][k16][vT]  ~6.7 MB.
  _Float16* wsH = (_Float16*)d_ws;
  _Float16* wT  = wsH;
  const size_t CH  = (size_t)C * H;
  const size_t BTH = (size_t)Bsz * T * H;
  _Float16* q16 = wsH + 3 * CH;
  _Float16* k16 = q16 + BTH;
  _Float16* vT  = k16 + BTH;

  wconvT_kernel<<<(int)(CH / 256), 256, 0, stream>>>(Wq, wT + 0 * CH);
  wconvT_kernel<<<(int)(CH / 256), 256, 0, stream>>>(Wk, wT + 1 * CH);
  wconvT_kernel<<<(int)(CH / 256), 256, 0, stream>>>(Wv, wT + 2 * CH);

  proj_kernel<<<(Bsz * T) / 64, 128, 0, stream>>>(x, wT, q16, k16, vT);

  attn_kernel<<<Bsz * (T / 64), 128, 0, stream>>>(q16, k16, vT, out);
}
